// MODEL_38929583571472
// MI455X (gfx1250) — compile-verified, matrix operand order checked by reading, not run on hardware
//
#include <hip/hip_runtime.h>
#include <hip/hip_bf16.h>

// ---------------- problem constants ----------------
#define Bv      8
#define Cc      384
#define Hh      96
#define Wcols   96
#define NHh     12
#define WINw    8
#define SHIFTs  4
#define HIDh    1536
#define Nn      64           // tokens per window
#define HDd     32           // head dim
#define NWIN    144          // 12*12 windows per image
#define BWIN    (Bv*NWIN)    // 1152
#define TOK     (Bv*Hh*Wcols) // 73728 tokens
#define SCALEf  0.17677669529663687f  // 32^-0.5
#define EPSf    1e-5f

typedef __attribute__((ext_vector_type(16))) _Float16 v16h;
typedef __attribute__((ext_vector_type(8)))  _Float16 v8h;
typedef __attribute__((ext_vector_type(8)))  float    v8f;

__device__ __forceinline__ v16h pack16(v8h lo, v8h hi) {
    v16h r;
#pragma unroll
    for (int i = 0; i < 8; ++i) { r[i] = lo[i]; r[i + 8] = hi[i]; }
    return r;
}

// shift-mask region id for a (shifted-image) coordinate
__device__ __forceinline__ int grp(int p) {
    return (p < Hh - WINw) ? 0 : ((p < Hh - SHIFTs) ? 1 : 2);
}

// ---------------- fp32 -> f16 weight conversion ----------------
__global__ __launch_bounds__(256) void f32_to_f16(const float* __restrict__ s,
                                                  _Float16* __restrict__ d, int n) {
    int i = blockIdx.x * 256 + threadIdx.x;
    if (i < n) d[i] = (_Float16)s[i];
}

// ---------------- LayerNorm (+ optional shift & window partition) ----------------
// one 128-thread block per token; C = 384 = 128*3
template <bool SHIFTED>
__global__ __launch_bounds__(128)
void ln_kernel(const float* __restrict__ x, const float* __restrict__ gw,
               const float* __restrict__ gb, _Float16* __restrict__ out) {
    __shared__ float red[8];
    int tok = blockIdx.x;
    const float* xp = x + (size_t)tok * Cc;
    float vals[3], s0 = 0.f, s1 = 0.f;
#pragma unroll
    for (int i = 0; i < 3; ++i) {
        float t = xp[threadIdx.x + i * 128];
        vals[i] = t; s0 += t; s1 += t * t;
    }
#pragma unroll
    for (int d = 1; d < 32; d <<= 1) {
        s0 += __shfl_xor(s0, d, 32);
        s1 += __shfl_xor(s1, d, 32);
    }
    int wid = threadIdx.x >> 5;
    if ((threadIdx.x & 31) == 0) { red[wid] = s0; red[4 + wid] = s1; }
    __syncthreads();
    float m0 = red[0] + red[1] + red[2] + red[3];
    float m1 = red[4] + red[5] + red[6] + red[7];
    float mu  = m0 * (1.0f / Cc);
    float inv = rsqrtf(m1 * (1.0f / Cc) - mu * mu + EPSf);
    size_t drow;
    if (SHIFTED) {
        int bw  = tok / (Hh * Wcols);
        int rem = tok - bw * (Hh * Wcols);
        int h = rem / Wcols, w = rem - (rem / Wcols) * Wcols;
        int h2 = (h + Hh - SHIFTs) % Hh;
        int w2 = (w + Wcols - SHIFTs) % Wcols;
        drow = (((size_t)bw * NWIN + (h2 >> 3) * 12 + (w2 >> 3)) * Nn + (h2 & 7) * 8 + (w2 & 7));
    } else {
        drow = tok;
    }
    _Float16* dp = out + drow * Cc;
#pragma unroll
    for (int i = 0; i < 3; ++i) {
        int c = threadIdx.x + i * 128;
        dp[c] = (_Float16)((vals[i] - mu) * inv * gw[c] + gb[c]);
    }
}

// ---------------- generic WMMA GEMM: out = A(MxK) @ Wt(NoutxK)^T ----------------
// block = 256 threads (8 waves); tile = 32 rows x 128 cols; A staged in LDS via
// CDNA5 async global->LDS loads (ASYNCcnt). Two accumulators per wave share the
// B fragment (2 WMMAs per k-step).
// EPI: 0 = qkv scatter (+bias, q*scale)  1 = proj (+bias, win-reverse+roll, +resid -> f32)
//      2 = fc1 (+bias, exact gelu -> f16 row-major)  3 = fc2 (+bias, +resid -> f32 d_out)
template <int EPI>
__global__ __launch_bounds__(256)
void gemm_wmma(const _Float16* __restrict__ A, const _Float16* __restrict__ Wt,
               const float* __restrict__ bias, const float* __restrict__ resid,
               float* __restrict__ outF, _Float16* __restrict__ oH0,
               _Float16* __restrict__ oH1, _Float16* __restrict__ oH2,
               int M, int K, int Nout) {
    extern __shared__ __align__(16) char smem[];
    _Float16* sA = (_Float16*)smem;                 // 32 x K halves
    int row0 = blockIdx.x * 32;

    // ---- async stage A rows [row0, row0+32) into LDS (contiguous region) ----
    {
        const _Float16* gbase = A + (size_t)row0 * K;
        unsigned long long gb64 = (unsigned long long)(uintptr_t)gbase;
        unsigned ldsBase = (unsigned)(uintptr_t)sA; // low 32 bits = LDS offset
        unsigned total = (unsigned)(32 * K * 2);    // bytes
        for (unsigned off = threadIdx.x * 16u; off < total; off += 256u * 16u) {
            unsigned l = ldsBase + off;
            asm volatile("global_load_async_to_lds_b128 %0, %1, %2 offset:0"
                         :: "v"(l), "v"(off), "s"(gb64)
                         : "memory");
        }
        asm volatile("s_wait_asynccnt 0x0" ::: "memory");
    }
    __syncthreads();

    int wave = threadIdx.x >> 5, lane = threadIdx.x & 31;
    int laneHi = lane >> 4, lane16 = lane & 15;
    int col = blockIdx.y * 128 + wave * 16 + lane16;
    const _Float16* wrow = Wt + (size_t)col * K;

    v8f acc0 = {}, acc1 = {};
    for (int kk = 0; kk < K; kk += 32) {
        int k0 = kk + laneHi * 8;
        v8h blo = *(const v8h*)&wrow[k0];
        v8h bhi = *(const v8h*)&wrow[k0 + 16];
        __builtin_prefetch(&wrow[k0 + 64], 0, 1);   // global_prefetch_b8 into L2
        v16h bv = pack16(blo, bhi);
        v16h av0 = pack16(*(const v8h*)&sA[lane16 * K + k0],
                          *(const v8h*)&sA[lane16 * K + k0 + 16]);
        v16h av1 = pack16(*(const v8h*)&sA[(16 + lane16) * K + k0],
                          *(const v8h*)&sA[(16 + lane16) * K + k0 + 16]);
        acc0 = __builtin_amdgcn_wmma_f32_16x16x32_f16(false, av0, false, bv,
                                                      (short)0, acc0, false, false);
        acc1 = __builtin_amdgcn_wmma_f32_16x16x32_f16(false, av1, false, bv,
                                                      (short)0, acc1, false, false);
    }

    float bval = bias ? bias[col] : 0.0f;
#pragma unroll
    for (int rb = 0; rb < 2; ++rb) {
        const v8f& acc = rb ? acc1 : acc0;
#pragma unroll
        for (int j = 0; j < 8; ++j) {
            int row = row0 + rb * 16 + j + laneHi * 8; // C/D layout
            float v = acc[j] + bval;
            if (EPI == 0) {
                int which = col / Cc;               // 0=q 1=k 2=v
                int cc = col - which * Cc;
                int head = cc >> 5, d = cc & 31;
                if (which == 0) v *= SCALEf;
                int win = row >> 6, n = row & 63;
                _Float16* dp = (which == 0) ? oH0 : ((which == 1) ? oH1 : oH2);
                dp[(((size_t)win * NHh + head) * Nn + n) * HDd + d] = (_Float16)v;
            } else if (EPI == 1) {
                int win = row >> 6, n = row & 63;
                int bw = win / NWIN, wrem = win - bw * NWIN;
                int wh = wrem / 12, wwi = wrem - wh * 12;
                int ii = n >> 3, jj = n & 7;
                int h = (wh * 8 + ii + SHIFTs) % Hh;       // window-reverse + roll(+SHIFT)
                int w = (wwi * 8 + jj + SHIFTs) % Wcols;
                size_t addr = (((size_t)bw * Hh + h) * Wcols + w) * Cc + col;
                outF[addr] = v + resid[addr];              // + shortcut (raw x view)
            } else if (EPI == 2) {
                float g = 0.5f * v * (1.0f + erff(v * 0.70710678118f)); // exact gelu
                oH0[(size_t)row * Nout + col] = (_Float16)g;
            } else {
                size_t addr = (size_t)row * Nout + col;
                outF[addr] = v + resid[addr];              // + xo residual -> d_out
            }
        }
    }
}

// ---------------- windowed attention: one block (4 waves) per (window, head) -----
__global__ __launch_bounds__(128)
void attn_kernel(const _Float16* __restrict__ q, const _Float16* __restrict__ k,
                 const _Float16* __restrict__ vv, const float* __restrict__ rpb,
                 _Float16* __restrict__ out) {
    __shared__ _Float16 sP[64 * 64];    // softmax(P) staged for layout change C->A
    __shared__ _Float16 sVt[32 * 64];   // V transposed: sVt[d][m]
    int bh = blockIdx.x;
    int head = bh % NHh;
    int win  = bh / NHh;
    int wi = win % NWIN;
    int wh = wi / 12, wwp = wi - wh * 12;
    const _Float16* Q  = q  + (size_t)bh * Nn * HDd;
    const _Float16* Km = k  + (size_t)bh * Nn * HDd;
    const _Float16* Vm = vv + (size_t)bh * Nn * HDd;

    for (int idx = threadIdx.x; idx < Nn * HDd; idx += 128) {
        int m = idx >> 5, d = idx & 31;
        sVt[d * 64 + m] = Vm[idx];
    }

    int wave = threadIdx.x >> 5, lane = threadIdx.x & 31;
    int laneHi = lane >> 4, lane16 = lane & 15;
    int m0 = wave * 16;               // this wave owns query rows [m0, m0+16)
    int k0 = laneHi * 8;

    // S = (Q*scale) K^T : HD=32 => one wmma per 16x16 tile
    v16h av = pack16(*(const v8h*)&Q[(m0 + lane16) * HDd + k0],
                     *(const v8h*)&Q[(m0 + lane16) * HDd + k0 + 16]);
    v8f s[4];
#pragma unroll
    for (int tn = 0; tn < 4; ++tn) {
        v16h bv = pack16(*(const v8h*)&Km[(tn * 16 + lane16) * HDd + k0],
                         *(const v8h*)&Km[(tn * 16 + lane16) * HDd + k0 + 16]);
        v8f z = {};
        s[tn] = __builtin_amdgcn_wmma_f32_16x16x32_f16(false, av, false, bv,
                                                       (short)0, z, false, false);
    }

    // bias + shift-mask + row softmax (rows split across 16-lane halves)
#pragma unroll
    for (int j = 0; j < 8; ++j) {
        int row = m0 + j + laneHi * 8;
        int gr = grp(wh * 8 + (row >> 3)) * 3 + grp(wwp * 8 + (row & 7));
        float valv[4];
#pragma unroll
        for (int tn = 0; tn < 4; ++tn) {
            int colm = tn * 16 + lane16;
            int dh = (row >> 3) - (colm >> 3) + 7;
            int dw = (row & 7) - (colm & 7) + 7;
            float bias = rpb[(dh * 15 + dw) * NHh + head];
            int gc = grp(wh * 8 + (colm >> 3)) * 3 + grp(wwp * 8 + (colm & 7));
            valv[tn] = s[tn][j] + bias + ((gr == gc) ? 0.f : -100.f);
        }
        float mx = fmaxf(fmaxf(valv[0], valv[1]), fmaxf(valv[2], valv[3]));
#pragma unroll
        for (int d = 1; d < 16; d <<= 1) mx = fmaxf(mx, __shfl_xor(mx, d, 32));
        float sum = 0.f;
#pragma unroll
        for (int tn = 0; tn < 4; ++tn) { valv[tn] = __expf(valv[tn] - mx); sum += valv[tn]; }
#pragma unroll
        for (int d = 1; d < 16; d <<= 1) sum += __shfl_xor(sum, d, 32);
        float invs = 1.0f / sum;
#pragma unroll
        for (int tn = 0; tn < 4; ++tn)
            sP[row * 64 + tn * 16 + lane16] = (_Float16)(valv[tn] * invs);
    }
    __syncthreads();

    // O = P @ V  (K=64 -> two k-steps; 2 n-tiles of 16)
    v8f o0 = {}, o1 = {};
    for (int ks = 0; ks < 64; ks += 32) {
        int kk0 = ks + laneHi * 8;
        v16h pa = pack16(*(const v8h*)&sP[(m0 + lane16) * 64 + kk0],
                         *(const v8h*)&sP[(m0 + lane16) * 64 + kk0 + 16]);
        v16h b0 = pack16(*(const v8h*)&sVt[lane16 * 64 + kk0],
                         *(const v8h*)&sVt[lane16 * 64 + kk0 + 16]);
        v16h b1 = pack16(*(const v8h*)&sVt[(16 + lane16) * 64 + kk0],
                         *(const v8h*)&sVt[(16 + lane16) * 64 + kk0 + 16]);
        o0 = __builtin_amdgcn_wmma_f32_16x16x32_f16(false, pa, false, b0, (short)0, o0, false, false);
        o1 = __builtin_amdgcn_wmma_f32_16x16x32_f16(false, pa, false, b1, (short)0, o1, false, false);
    }
#pragma unroll
    for (int j = 0; j < 8; ++j) {
        int row = m0 + j + laneHi * 8;
        size_t base = ((size_t)win * Nn + row) * Cc + head * HDd;
        out[base + lane16]      = (_Float16)o0[j];
        out[base + 16 + lane16] = (_Float16)o1[j];
    }
}

// ---------------- launcher ----------------
extern "C" void kernel_launch(void* const* d_in, const int* in_sizes, int n_in,
                              void* d_out, int out_size, void* d_ws, size_t ws_size,
                              hipStream_t stream) {
    (void)in_sizes; (void)n_in; (void)out_size; (void)ws_size;
    const float* x      = (const float*)d_in[0];
    const float* qkv_w  = (const float*)d_in[1];
    const float* qkv_b  = (const float*)d_in[2];
    const float* rpb    = (const float*)d_in[3];
    const float* proj_w = (const float*)d_in[4];
    const float* proj_b = (const float*)d_in[5];
    const float* n1w    = (const float*)d_in[6];
    const float* n1b    = (const float*)d_in[7];
    const float* n2w    = (const float*)d_in[8];
    const float* n2b    = (const float*)d_in[9];
    const float* fc1w   = (const float*)d_in[10];
    const float* fc1b   = (const float*)d_in[11];
    const float* fc2w   = (const float*)d_in[12];
    const float* fc2b   = (const float*)d_in[13];

    char* ws = (char*)d_ws;
    size_t off = 0;
    auto take = [&](size_t bytes) -> char* {
        char* p = ws + off;
        off = (off + bytes + 255) & ~(size_t)255;
        return p;
    };
    _Float16* xw     = (_Float16*)take((size_t)TOK * Cc * 2);     // LN1+shift+partition
    _Float16* qf     = (_Float16*)take((size_t)TOK * Cc * 2);     // (B_,NH,N,HD)
    _Float16* kf     = (_Float16*)take((size_t)TOK * Cc * 2);
    _Float16* vf     = (_Float16*)take((size_t)TOK * Cc * 2);
    _Float16* ao     = (_Float16*)take((size_t)TOK * Cc * 2);     // attn out (B_,N,C)
    float*    xo     = (float*)   take((size_t)TOK * Cc * 4);     // residual stream
    _Float16* xn2    = (_Float16*)take((size_t)TOK * Cc * 2);     // LN2 out
    _Float16* h1     = (_Float16*)take((size_t)TOK * HIDh * 2);   // gelu(fc1)
    _Float16* qkvw16 = (_Float16*)take((size_t)3 * Cc * Cc * 2);
    _Float16* projw16= (_Float16*)take((size_t)Cc * Cc * 2);
    _Float16* fc1w16 = (_Float16*)take((size_t)HIDh * Cc * 2);
    _Float16* fc2w16 = (_Float16*)take((size_t)Cc * HIDh * 2);

    auto cvt = [&](const float* s, _Float16* d, int n) {
        hipLaunchKernelGGL(f32_to_f16, dim3((n + 255) / 256), dim3(256), 0, stream, s, d, n);
    };
    cvt(qkv_w, qkvw16, 3 * Cc * Cc);
    cvt(proj_w, projw16, Cc * Cc);
    cvt(fc1w, fc1w16, HIDh * Cc);
    cvt(fc2w, fc2w16, Cc * HIDh);

    hipLaunchKernelGGL((ln_kernel<true>), dim3(TOK), dim3(128), 0, stream, x, n1w, n1b, xw);

    hipLaunchKernelGGL((gemm_wmma<0>), dim3(TOK / 32, (3 * Cc) / 128), dim3(256),
                       (size_t)(32 * Cc * 2), stream,
                       xw, qkvw16, qkv_b, (const float*)nullptr,
                       (float*)nullptr, qf, kf, vf, TOK, Cc, 3 * Cc);

    hipLaunchKernelGGL(attn_kernel, dim3(BWIN * NHh), dim3(128), 0, stream,
                       qf, kf, vf, rpb, ao);

    hipLaunchKernelGGL((gemm_wmma<1>), dim3(TOK / 32, Cc / 128), dim3(256),
                       (size_t)(32 * Cc * 2), stream,
                       ao, projw16, proj_b, x, xo,
                       (_Float16*)nullptr, (_Float16*)nullptr, (_Float16*)nullptr,
                       TOK, Cc, Cc);

    hipLaunchKernelGGL((ln_kernel<false>), dim3(TOK), dim3(128), 0, stream, xo, n2w, n2b, xn2);

    hipLaunchKernelGGL((gemm_wmma<2>), dim3(TOK / 32, HIDh / 128), dim3(256),
                       (size_t)(32 * Cc * 2), stream,
                       xn2, fc1w16, fc1b, (const float*)nullptr,
                       (float*)nullptr, h1, (_Float16*)nullptr, (_Float16*)nullptr,
                       TOK, Cc, HIDh);

    hipLaunchKernelGGL((gemm_wmma<3>), dim3(TOK / 32, Cc / 128), dim3(256),
                       (size_t)(32 * HIDh * 2), stream,
                       h1, fc2w16, fc2b, xo, (float*)d_out,
                       (_Float16*)nullptr, (_Float16*)nullptr, (_Float16*)nullptr,
                       TOK, HIDh, Cc);
}
